// STE_Mamba_6897717477711
// MI455X (gfx1250) — compile-verified, matrix operand order checked by reading, not run on hardware
//
#include <hip/hip_runtime.h>
#include <hip/hip_bf16.h>

// ---------------------------------------------------------------------------
// Types for CDNA5 WMMA
// ---------------------------------------------------------------------------
typedef __bf16 bf16_t;
typedef bf16_t v16bf __attribute__((ext_vector_type(16)));
typedef float  v8f   __attribute__((ext_vector_type(8)));
using u16 = unsigned short;

// Pointer types for the async global->LDS builtins (arg type from clang diag:
// pointer to `int __attribute__((vector_size(16)))` in AS1 / AS3)
typedef int v4i __attribute__((vector_size(16)));
typedef __attribute__((address_space(1))) v4i* as1_v4i;
typedef __attribute__((address_space(3))) v4i* as3_v4i;

union FragU {
    uint4 q[2];
    v16bf v;
};

__device__ __forceinline__ u16 f2bf(float f) {
    unsigned int u = __builtin_bit_cast(unsigned int, f);
    unsigned int r = u + 0x7FFFu + ((u >> 16) & 1u);   // round-to-nearest-even
    return (u16)(r >> 16);
}

__device__ __forceinline__ float sigmoidf(float x) { return 1.0f / (1.0f + __expf(-x)); }
__device__ __forceinline__ float siluf(float x)    { return x * sigmoidf(x); }
__device__ __forceinline__ float softplusf(float x){ return (x > 20.0f) ? x : log1pf(__expf(x)); }

// Problem constants
#define MTOK   8192      // B*L tokens
#define DIMC   512
#define DINNER 1024
#define DSTATE 64
#define DTRANK 32
#define LSEQ   2048
#define NBATCH 4

// ---------------------------------------------------------------------------
// CDNA5 async global->LDS copy (GLOBAL_LOAD_ASYNC_TO_LDS_B128, ASYNCcnt)
// ---------------------------------------------------------------------------
__device__ __forceinline__ void cp_async_b128(const void* g, void* l) {
#if defined(__HIP_DEVICE_COMPILE__)
#if __has_builtin(__builtin_amdgcn_global_load_async_to_lds_b128)
    __builtin_amdgcn_global_load_async_to_lds_b128(
        (as1_v4i)(unsigned long long)g,
        (as3_v4i)(unsigned int)(unsigned long long)l,
        0, 0);
#else
    asm volatile("global_load_async_to_lds_b128 %0, %1, off"
                 :: "v"((unsigned int)(unsigned long long)l),
                    "v"((unsigned long long)g)
                 : "memory");
#endif
#else
    (void)g; (void)l;
#endif
}

__device__ __forceinline__ void wait_async3() {
#if defined(__HIP_DEVICE_COMPILE__)
#if __has_builtin(__builtin_amdgcn_s_wait_asynccnt)
    __builtin_amdgcn_s_wait_asynccnt(3);
#else
    asm volatile("s_wait_asynccnt 0x3" ::: "memory");
#endif
#endif
}

__device__ __forceinline__ void wait_async0() {
#if defined(__HIP_DEVICE_COMPILE__)
#if __has_builtin(__builtin_amdgcn_s_wait_asynccnt)
    __builtin_amdgcn_s_wait_asynccnt(0);
#else
    asm volatile("s_wait_asynccnt 0x0" ::: "memory");
#endif
#endif
}

// ---------------------------------------------------------------------------
// f32 -> bf16 repack (weights & activations)
// ---------------------------------------------------------------------------
__global__ void k_f32_to_bf16(const float* __restrict__ in, u16* __restrict__ out, long n) {
    long i = (long)blockIdx.x * blockDim.x + threadIdx.x;
    if (i < n) out[i] = f2bf(in[i]);
}

// ---------------------------------------------------------------------------
// LayerNorm over last dim (512), one block per token, writes bf16
// ---------------------------------------------------------------------------
__global__ void k_layernorm_bf16(const float* __restrict__ x,
                                 const float* __restrict__ w,
                                 const float* __restrict__ b,
                                 u16* __restrict__ out) {
    __shared__ float red[256];
    const int tid = threadIdx.x;
    const size_t base = (size_t)blockIdx.x * DIMC;
    float v0 = x[base + tid];
    float v1 = x[base + tid + 256];

    red[tid] = v0 + v1;
    __syncthreads();
    for (int s = 128; s > 0; s >>= 1) {
        if (tid < s) red[tid] += red[tid + s];
        __syncthreads();
    }
    const float mu = red[0] * (1.0f / DIMC);
    __syncthreads();

    float d0 = v0 - mu, d1 = v1 - mu;
    red[tid] = d0 * d0 + d1 * d1;
    __syncthreads();
    for (int s = 128; s > 0; s >>= 1) {
        if (tid < s) red[tid] += red[tid + s];
        __syncthreads();
    }
    const float rs = rsqrtf(red[0] * (1.0f / DIMC) + 1e-5f);

    out[base + tid]       = f2bf(d0 * rs * w[tid]       + b[tid]);
    out[base + tid + 256] = f2bf(d1 * rs * w[tid + 256] + b[tid + 256]);
}

// ---------------------------------------------------------------------------
// Generic GEMM: C[M,N](f32) = A[M,K](bf16) * W[N,K]^T(bf16) (+ bias[N])
// Block tile 128x64, 8 waves, each wave 32x32 via 2x2 v_wmma_f32_16x16x32_bf16
// Double-buffered LDS tiles staged with GLOBAL_LOAD_ASYNC_TO_LDS_B128.
// ---------------------------------------------------------------------------
#define TPAD 40   // padded LDS row stride in bf16 elems (80B, 16B aligned)

__device__ __forceinline__ v16bf load_frag(const u16* p) {
    FragU f;
    f.q[0] = *(const uint4*)(p);        // K offsets ks .. ks+7
    f.q[1] = *(const uint4*)(p + 16);   // K offsets ks+16 .. ks+23
    return f.v;
}

__global__ __launch_bounds__(256)
void k_gemm_bf16_wmma(const u16* __restrict__ A,
                      const u16* __restrict__ W,
                      const float* __restrict__ bias,
                      float* __restrict__ C,
                      int M, int N, int K) {
    __shared__ __align__(16) u16 sA[2][128 * TPAD];
    __shared__ __align__(16) u16 sB[2][64 * TPAD];

    const int tid  = threadIdx.x;
    const int m0   = blockIdx.x * 128;
    const int n0   = blockIdx.y * 64;
    const int wave = tid >> 5;
    const int lane = tid & 31;
    const int wm   = (wave & 3) * 32;   // 4 waves along M
    const int wn   = (wave >> 2) * 32;  // 2 waves along N
    const int lm   = lane & 15;
    const int lh   = lane >> 4;
    const int ks   = lh * 8;

    // Staging coordinates: thread moves 2x16B of A (rows r, r+64) + 1x16B of B
    const int sRow = tid >> 2;
    const int sCol = (tid & 3) * 8;
    int bRow = n0 + sRow;
    if (bRow >= N) bRow = N - 1;       // clamp: garbage cols never stored

    v8f acc[2][2];
    #pragma unroll
    for (int i = 0; i < 2; ++i)
        #pragma unroll
        for (int j = 0; j < 2; ++j)
            acc[i][j] = (v8f){0.f,0.f,0.f,0.f,0.f,0.f,0.f,0.f};

    // Prologue: stage K-tile 0 into buffer 0 (3 async ops per thread)
    cp_async_b128(A + (size_t)(m0 + sRow)      * K + sCol, &sA[0][ sRow       * TPAD + sCol]);
    cp_async_b128(A + (size_t)(m0 + sRow + 64) * K + sCol, &sA[0][(sRow + 64) * TPAD + sCol]);
    cp_async_b128(W + (size_t)bRow             * K + sCol, &sB[0][ sRow       * TPAD + sCol]);

    for (int k0 = 0; k0 < K; k0 += 32) {
        const int cur = (k0 >> 5) & 1;
        if (k0 + 32 < K) {
            const int nxt = cur ^ 1;
            const int kn  = k0 + 32;
            cp_async_b128(A + (size_t)(m0 + sRow)      * K + kn + sCol, &sA[nxt][ sRow       * TPAD + sCol]);
            cp_async_b128(A + (size_t)(m0 + sRow + 64) * K + kn + sCol, &sA[nxt][(sRow + 64) * TPAD + sCol]);
            cp_async_b128(W + (size_t)bRow             * K + kn + sCol, &sB[nxt][ sRow       * TPAD + sCol]);
            if (k0 + 64 < K)
                __builtin_prefetch(A + (size_t)(m0 + sRow) * K + k0 + 64 + sCol, 0, 1);
            wait_async3();   // tile `cur` complete; tile `nxt` may be in flight
        } else {
            wait_async0();
        }
        __syncthreads();

        const u16* bufA = sA[cur];
        const u16* bufB = sB[cur];
        v16bf a0 = load_frag(&bufA[(wm + lm)      * TPAD + ks]);
        v16bf a1 = load_frag(&bufA[(wm + lm + 16) * TPAD + ks]);
        v16bf b0 = load_frag(&bufB[(wn + lm)      * TPAD + ks]);
        v16bf b1 = load_frag(&bufB[(wn + lm + 16) * TPAD + ks]);

        acc[0][0] = __builtin_amdgcn_wmma_f32_16x16x32_bf16(false, a0, false, b0, (short)0, acc[0][0], false, false);
        acc[0][1] = __builtin_amdgcn_wmma_f32_16x16x32_bf16(false, a0, false, b1, (short)0, acc[0][1], false, false);
        acc[1][0] = __builtin_amdgcn_wmma_f32_16x16x32_bf16(false, a1, false, b0, (short)0, acc[1][0], false, false);
        acc[1][1] = __builtin_amdgcn_wmma_f32_16x16x32_bf16(false, a1, false, b1, (short)0, acc[1][1], false, false);

        __syncthreads();
    }

    // Store: D layout -> lane holds (m = r + 8*lh, n = lm) of each 16x16 tile
    #pragma unroll
    for (int i = 0; i < 2; ++i) {
        #pragma unroll
        for (int j = 0; j < 2; ++j) {
            int n = n0 + wn + 16 * j + lm;
            if (n < N) {
                float bv = bias ? bias[n] : 0.0f;
                #pragma unroll
                for (int r = 0; r < 8; ++r) {
                    int m = m0 + wm + 16 * i + r + 8 * lh;
                    C[(size_t)m * N + n] = acc[i][j][r] + bv;
                }
            }
        }
    }
}

// ---------------------------------------------------------------------------
// Causal depthwise conv (D_CONV=4) + SiLU over xin = xz[:, :1024]
// ---------------------------------------------------------------------------
__global__ void k_conv_silu(const float* __restrict__ xz,
                            const float* __restrict__ cw,
                            const float* __restrict__ cb,
                            float* __restrict__ u,
                            u16* __restrict__ ubf) {
    long idx = (long)blockIdx.x * blockDim.x + threadIdx.x;
    if (idx >= (long)MTOK * DINNER) return;
    int e = (int)(idx & (DINNER - 1));
    long t = idx >> 10;                 // token = b*L + l
    int l  = (int)(t & (LSEQ - 1));
    float s = cb[e];
    #pragma unroll
    for (int k = 0; k < 4; ++k) {
        int lk = l - 3 + k;
        if (lk >= 0)
            s += cw[e * 4 + k] * xz[(size_t)(t - 3 + k) * (2 * DINNER) + e];
    }
    float sv = siluf(s);
    u[idx]   = sv;
    ubf[idx] = f2bf(sv);
}

// ---------------------------------------------------------------------------
// Extract dt (first 32 cols of xdb[M,160]) to bf16
// ---------------------------------------------------------------------------
__global__ void k_dt_extract(const float* __restrict__ xdb, u16* __restrict__ dtbf) {
    long idx = (long)blockIdx.x * blockDim.x + threadIdx.x;
    if (idx >= (long)MTOK * DTRANK) return;
    long row = idx >> 5;
    int  c   = (int)(idx & 31);
    dtbf[idx] = f2bf(xdb[row * (DTRANK + 2 * DSTATE) + c]);
}

// ---------------------------------------------------------------------------
// Selective scan, fused: softplus(dt_proj + b), recurrence over L, gating
//   y = (scan_y + u*D) * silu(z) -> bf16 (input to out_proj GEMM)
// One thread per (batch, channel); 64 f32 states in registers.
// ---------------------------------------------------------------------------
__global__ __launch_bounds__(256)
void k_selective_scan(const float* __restrict__ xdb,     // [M,160]: dt|B|C
                      const float* __restrict__ dtpo,    // [M,1024]
                      const float* __restrict__ dt_b,    // [1024]
                      const float* __restrict__ u,       // [M,1024]
                      const float* __restrict__ xz,      // [M,2048] (z half)
                      const float* __restrict__ A_log,   // [1024,64]
                      const float* __restrict__ Dp,      // [1024]
                      u16* __restrict__ ybf) {           // [M,1024]
    __shared__ float Bs[DSTATE];
    __shared__ float Cs[DSTATE];

    const int tid = threadIdx.x;
    const int b   = blockIdx.x >> 2;            // 4 blocks per batch
    const int e   = (blockIdx.x & 3) * 256 + tid;

    float a[DSTATE];
    float h[DSTATE];
    #pragma unroll
    for (int n = 0; n < DSTATE; ++n) {
        a[n] = -__expf(A_log[(size_t)e * DSTATE + n]);
        h[n] = 0.0f;
    }
    const float dbias = dt_b[e];
    const float dcoef = Dp[e];

    for (int t = 0; t < LSEQ; ++t) {
        const size_t tok = (size_t)b * LSEQ + t;
        if (tid < DSTATE)
            Bs[tid] = xdb[tok * (DTRANK + 2 * DSTATE) + DTRANK + tid];
        else if (tid < 2 * DSTATE)
            Cs[tid - DSTATE] = xdb[tok * (DTRANK + 2 * DSTATE) + DTRANK + DSTATE + (tid - DSTATE)];
        __syncthreads();

        const float delta = softplusf(dtpo[tok * DINNER + e] + dbias);
        const float ut    = u[tok * DINNER + e];
        const float du    = delta * ut;

        float y = 0.0f;
        #pragma unroll
        for (int n = 0; n < DSTATE; ++n) {
            float dA = __expf(delta * a[n]);
            h[n] = fmaf(dA, h[n], du * Bs[n]);
            y    = fmaf(h[n], Cs[n], y);
        }

        const float zv = xz[tok * (2 * DINNER) + DINNER + e];
        const float yo = (y + ut * dcoef) * siluf(zv);
        ybf[tok * DINNER + e] = f2bf(yo);
        __syncthreads();
    }
}

// ---------------------------------------------------------------------------
// Elementwise kernels
// ---------------------------------------------------------------------------
__global__ void k_add(const float* __restrict__ a, const float* __restrict__ bsrc,
                      float* __restrict__ o, long n) {
    long i = (long)blockIdx.x * blockDim.x + threadIdx.x;
    if (i < n) o[i] = a[i] + bsrc[i];
}

__global__ void k_gelu_bf16(const float* __restrict__ in, u16* __restrict__ out, long n) {
    long i = (long)blockIdx.x * blockDim.x + threadIdx.x;
    if (i < n) {
        float v = in[i];
        out[i] = f2bf(0.5f * v * (1.0f + erff(v * 0.70710678118654752f)));
    }
}

// ---------------------------------------------------------------------------
// Host orchestration
// ---------------------------------------------------------------------------
extern "C" void kernel_launch(void* const* d_in, const int* in_sizes, int n_in,
                              void* d_out, int out_size, void* d_ws, size_t ws_size,
                              hipStream_t stream) {
    const float* x        = (const float*)d_in[0];
    const float* ln1_w    = (const float*)d_in[1];
    const float* ln1_b    = (const float*)d_in[2];
    const float* in_pw    = (const float*)d_in[3];
    const float* conv_w   = (const float*)d_in[4];
    const float* conv_b   = (const float*)d_in[5];
    const float* x_pw     = (const float*)d_in[6];
    const float* dt_pw    = (const float*)d_in[7];
    const float* dt_pb    = (const float*)d_in[8];
    const float* A_log    = (const float*)d_in[9];
    const float* Dvec     = (const float*)d_in[10];
    const float* out_pw   = (const float*)d_in[11];
    const float* ln2_w    = (const float*)d_in[12];
    const float* ln2_b    = (const float*)d_in[13];
    const float* fc1_w    = (const float*)d_in[14];
    const float* fc1_b    = (const float*)d_in[15];
    const float* fc2_w    = (const float*)d_in[16];
    const float* fc2_b    = (const float*)d_in[17];
    float* out = (float*)d_out;

    char* ws = (char*)d_ws;
    size_t off = 0;
    auto take = [&](size_t bytes) -> char* {
        char* p = ws + off;
        off = (off + bytes + 255) & ~(size_t)255;
        return p;
    };

    const size_t M = MTOK;
    // bf16 weights
    u16* wb_in  = (u16*)take((size_t)2048 * 512  * 2);
    u16* wb_x   = (u16*)take((size_t)160  * 1024 * 2);
    u16* wb_dt  = (u16*)take((size_t)1024 * 32   * 2);
    u16* wb_out = (u16*)take((size_t)512  * 1024 * 2);
    u16* wb_f1  = (u16*)take((size_t)2048 * 512  * 2);
    u16* wb_f2  = (u16*)take((size_t)512  * 2048 * 2);
    // activations
    u16*   h1bf = (u16*)take(M * 512  * 2);
    float* xz   = (float*)take(M * 2048 * 4);
    float* u_f  = (float*)take(M * 1024 * 4);
    u16*   ubf  = (u16*)take(M * 1024 * 2);
    float* xdb  = (float*)take(M * 160  * 4);
    u16*   dtbf = (u16*)take(M * 32   * 2);
    float* dtpo = (float*)take(M * 1024 * 4);
    float* oprj = (float*)take(M * 512  * 4);
    float* x2   = (float*)take(M * 512  * 4);
    // aliases over dead buffers
    u16*   ybf   = ubf;            // free after x_proj GEMM
    u16*   h2bf  = h1bf;           // free after in_proj GEMM
    float* fc1o  = xz;             // free after scan (z consumed)
    u16*   ff1bf = (u16*)u_f;      // free after scan
    float* fc2o  = oprj;           // free after residual add

    const int T = 256;
    auto blks = [](long n, int t) { return (unsigned)((n + t - 1) / t); };

    // 1) repack weights to bf16
    k_f32_to_bf16<<<blks(2048L*512, T),  T, 0, stream>>>(in_pw,  wb_in,  2048L*512);
    k_f32_to_bf16<<<blks(160L*1024, T),  T, 0, stream>>>(x_pw,   wb_x,   160L*1024);
    k_f32_to_bf16<<<blks(1024L*32, T),   T, 0, stream>>>(dt_pw,  wb_dt,  1024L*32);
    k_f32_to_bf16<<<blks(512L*1024, T),  T, 0, stream>>>(out_pw, wb_out, 512L*1024);
    k_f32_to_bf16<<<blks(2048L*512, T),  T, 0, stream>>>(fc1_w,  wb_f1,  2048L*512);
    k_f32_to_bf16<<<blks(512L*2048, T),  T, 0, stream>>>(fc2_w,  wb_f2,  512L*2048);

    // 2) LN1 -> h1 (bf16)
    k_layernorm_bf16<<<(unsigned)M, 256, 0, stream>>>(x, ln1_w, ln1_b, h1bf);

    // 3) in_proj: xz[M,2048] = h1 * W_in^T
    k_gemm_bf16_wmma<<<dim3(M/128, 2048/64), 256, 0, stream>>>(h1bf, wb_in, nullptr, xz, (int)M, 2048, 512);

    // 4) conv + silu -> u (f32 + bf16)
    k_conv_silu<<<blks(M*1024L, T), T, 0, stream>>>(xz, conv_w, conv_b, u_f, ubf);

    // 5) x_proj: xdb[M,160] = u * W_x^T
    k_gemm_bf16_wmma<<<dim3(M/128, (160+63)/64), 256, 0, stream>>>(ubf, wb_x, nullptr, xdb, (int)M, 160, 1024);

    // 6) dt slice -> bf16
    k_dt_extract<<<blks(M*32L, T), T, 0, stream>>>(xdb, dtbf);

    // 7) dt_proj: dtpo[M,1024] = dt * W_dt^T
    k_gemm_bf16_wmma<<<dim3(M/128, 1024/64), 256, 0, stream>>>(dtbf, wb_dt, nullptr, dtpo, (int)M, 1024, 32);

    // 8) selective scan (fused softplus + gate) -> ybf
    k_selective_scan<<<NBATCH * (DINNER/256), 256, 0, stream>>>(xdb, dtpo, dt_pb, u_f, xz, A_log, Dvec, ybf);

    // 9) out_proj: oprj[M,512] = y * W_out^T
    k_gemm_bf16_wmma<<<dim3(M/128, 512/64), 256, 0, stream>>>(ybf, wb_out, nullptr, oprj, (int)M, 512, 1024);

    // 10) residual: x2 = x + mamba_out
    k_add<<<blks(M*512L, T), T, 0, stream>>>(x, oprj, x2, M*512L);

    // 11) LN2 -> h2 (bf16)
    k_layernorm_bf16<<<(unsigned)M, 256, 0, stream>>>(x2, ln2_w, ln2_b, h2bf);

    // 12) fc1: fc1o[M,2048] = h2 * W_f1^T + b1
    k_gemm_bf16_wmma<<<dim3(M/128, 2048/64), 256, 0, stream>>>(h2bf, wb_f1, fc1_b, fc1o, (int)M, 2048, 512);

    // 13) gelu -> bf16
    k_gelu_bf16<<<blks(M*2048L, T), T, 0, stream>>>(fc1o, ff1bf, M*2048L);

    // 14) fc2: fc2o[M,512] = ff * W_f2^T + b2
    k_gemm_bf16_wmma<<<dim3(M/128, 512/64), 256, 0, stream>>>(ff1bf, wb_f2, fc2_b, fc2o, (int)M, 512, 2048);

    // 15) out = x2 + ff
    k_add<<<blks(M*512L, T), T, 0, stream>>>(x2, fc2o, out, M*512L);
}
